// Memory_65034394796571
// MI455X (gfx1250) — compile-verified, matrix operand order checked by reading, not run on hardware
//
#include <hip/hip_runtime.h>
#include <hip/hip_bf16.h>
#include <stdint.h>

// ---------------- problem constants ----------------
constexpr int H = 148, W = 300;
constexpr int HW = H * W;          // 44400 pixels (divisible by 16)
constexpr int D  = 128;            // channels
constexpr int M  = 2048;           // memory slots
constexpr int CHUNK = 512;         // score column chunk
constexpr int NCHUNKS = (HW + CHUNK - 1) / CHUNK;   // 87
constexpr int PTILES = (HW + 255) / 256;            // 174 pixel macro-tiles

// ---------------- WMMA plumbing ----------------
typedef __bf16 v16bf __attribute__((ext_vector_type(16)));
typedef float  v8f   __attribute__((ext_vector_type(8)));

union Frag {
  uint4 q[2];     // 32 bytes = 16 bf16 elements
  v16bf v;
};

static __device__ __forceinline__ v8f vzero() {
  v8f z = {0.f,0.f,0.f,0.f,0.f,0.f,0.f,0.f};
  return z;
}

static __device__ __forceinline__ v8f wmma_bf16(const Frag& a, const Frag& b, v8f c) {
  // D = A(16x32) * B(32x16) + C, fp32 accumulate
  return __builtin_amdgcn_wmma_f32_16x16x32_bf16(
      false, a.v, false, b.v, (short)0, c, false, false);
}

static __device__ __forceinline__ unsigned short f2bf(float f) {
  unsigned int u = __float_as_uint(f);
  unsigned int r = (u + 0x7FFFu + ((u >> 16) & 1u)) >> 16;   // RNE
  return (unsigned short)r;
}

// ---------------- normalization ----------------
__global__ void k_norm_query(const float* __restrict__ x,
                             float* __restrict__ qn,
                             unsigned short* __restrict__ q2d_bf) {
  int p = blockIdx.x * blockDim.x + threadIdx.x;
  if (p >= HW) return;
  float ss = 0.f;
  for (int d = 0; d < D; ++d) { float v = x[(size_t)d*HW + p]; ss += v*v; }
  float inv = 1.f / fmaxf(sqrtf(ss), 1e-12f);
  for (int d = 0; d < D; ++d) {
    float v = x[(size_t)d*HW + p] * inv;
    qn[(size_t)d*HW + p] = v;                 // channel-major fp32 (reused 3x)
    q2d_bf[(size_t)p*D + d] = f2bf(v);        // pixel-major bf16 (GEMM operand)
  }
}

__global__ void k_norm_keys(const float* __restrict__ keys,
                            unsigned short* __restrict__ kn_bf,
                            float* __restrict__ out_keys) {
  int m = blockIdx.x * blockDim.x + threadIdx.x;
  if (m >= M) return;
  float ss = 0.f;
  for (int d = 0; d < D; ++d) { float v = keys[(size_t)m*D + d]; ss += v*v; }
  float inv = 1.f / fmaxf(sqrtf(ss), 1e-6f);  // folds cos-denominator clamp
  for (int d = 0; d < D; ++d) {
    float v = keys[(size_t)m*D + d];
    kn_bf[(size_t)m*D + d] = f2bf(v * inv);
    out_keys[(size_t)m*D + d] = v;            // updated_memory = keys
  }
}

// ---------------- score pass A: per-row sum of exp(cos-1) ----------------
__global__ __launch_bounds__(128)
void k_score_stats(const unsigned short* __restrict__ kn,
                   const unsigned short* __restrict__ q2d,
                   float* __restrict__ rsum_part) {
  const int lane = threadIdx.x & 31, wave = threadIdx.x >> 5;
  const int chunk = blockIdx.x;
  const int m0 = blockIdx.y * 64 + wave * 16;
  const int nstart = chunk * CHUNK;
  const int nend = (nstart + CHUNK < HW) ? (nstart + CHUNK) : HW;
  const int ln = lane & 15, kbase = (lane >> 4) * 8;
  const unsigned short* arow = kn + (size_t)(m0 + ln) * D;

  float rs[8];
  #pragma unroll
  for (int r = 0; r < 8; ++r) rs[r] = 0.f;

  for (int n0 = nstart; n0 < nend; n0 += 16) {
    const unsigned short* brow = q2d + (size_t)(n0 + ln) * D;
    v8f acc = vzero();
    #pragma unroll
    for (int kb = 0; kb < D; kb += 32) {
      Frag A, B;
      A.q[0] = *(const uint4*)(arow + kb + kbase);
      A.q[1] = *(const uint4*)(arow + kb + kbase + 16);
      B.q[0] = *(const uint4*)(brow + kb + kbase);
      B.q[1] = *(const uint4*)(brow + kb + kbase + 16);
      acc = wmma_bf16(A, B, acc);
    }
    #pragma unroll
    for (int r = 0; r < 8; ++r) rs[r] += __expf(acc[r] - 1.0f);
  }
  // reduce over the 16 n-lanes inside each half-wave (xor masks stay in-half)
  #pragma unroll
  for (int r = 0; r < 8; ++r)
    for (int off = 1; off < 16; off <<= 1)
      rs[r] += __shfl_xor(rs[r], off, 32);
  if (ln == 0) {
    int mbase = m0 + 8 * (lane >> 4);
    #pragma unroll
    for (int r = 0; r < 8; ++r)
      rsum_part[(size_t)(mbase + r) * NCHUNKS + chunk] = rs[r];
  }
}

__global__ void k_bias(const float* __restrict__ rsum_part, float* __restrict__ bias) {
  int m = blockIdx.x * blockDim.x + threadIdx.x;
  if (m >= M) return;
  float s = 0.f;
  for (int c = 0; c < NCHUNKS; ++c) s += rsum_part[(size_t)m*NCHUNKS + c];
  bias[m] = -(1.0f + __logf(fmaxf(s, 1e-30f)));  // argmax(sm) == argmax(cos+bias)
}

// ---------------- score pass B: online top-2 over m ----------------
static __device__ __forceinline__ void push2(float& v1, int& i1, float& v2, int& i2,
                                             float v, int i) {
  bool b1 = (v > v1) || (v == v1 && i < i1);
  if (b1) { v2 = v1; i2 = i1; v1 = v; i1 = i; }
  else {
    bool b2 = (v > v2) || (v == v2 && i < i2);
    if (b2) { v2 = v; i2 = i; }
  }
}

__global__ __launch_bounds__(128)
void k_top2(const unsigned short* __restrict__ kn,
            const unsigned short* __restrict__ q2d,
            const float* __restrict__ bias,
            int* __restrict__ top1, int* __restrict__ top2) {
  const int lane = threadIdx.x & 31, wave = threadIdx.x >> 5;
  const int gw = blockIdx.x * 4 + wave;
  if (gw >= HW / 16) return;
  const int n0 = gw * 16;
  const int ln = lane & 15, kbase = (lane >> 4) * 8;
  const unsigned short* brow = q2d + (size_t)(n0 + ln) * D;

  float v1 = -3.0e38f, v2 = -3.0e38f;
  int   i1 = 0x7fffffff, i2 = 0x7fffffff;

  for (int m0 = 0; m0 < M; m0 += 16) {
    const unsigned short* arow = kn + (size_t)(m0 + ln) * D;
    v8f acc = vzero();
    #pragma unroll
    for (int kb = 0; kb < D; kb += 32) {
      Frag A, B;
      A.q[0] = *(const uint4*)(arow + kb + kbase);
      A.q[1] = *(const uint4*)(arow + kb + kbase + 16);
      B.q[0] = *(const uint4*)(brow + kb + kbase);
      B.q[1] = *(const uint4*)(brow + kb + kbase + 16);
      acc = wmma_bf16(A, B, acc);
    }
    const int mbase = m0 + 8 * (lane >> 4);
    #pragma unroll
    for (int r = 0; r < 8; ++r) {
      float v = acc[r] + bias[mbase + r];
      push2(v1, i1, v2, i2, v, mbase + r);
    }
  }
  // merge the two half-wave m-subsets (lane ^ 16 holds same n, other 8 m's)
  float pv1 = __shfl_xor(v1, 16, 32), pv2 = __shfl_xor(v2, 16, 32);
  int   pi1 = __shfl_xor(i1, 16, 32), pi2 = __shfl_xor(i2, 16, 32);
  push2(v1, i1, v2, i2, pv1, pi1);
  push2(v1, i1, v2, i2, pv2, pi2);
  if (lane < 16) { top1[n0 + lane] = i1; top2[n0 + lane] = i2; }
}

// ---------------- losses ----------------
__global__ void k_init(float* accum) { if (threadIdx.x < 2) accum[threadIdx.x] = 0.f; }

__global__ __launch_bounds__(256)
void k_losses(const float* __restrict__ qn, const float* __restrict__ keys,
              const int* __restrict__ top1, const int* __restrict__ top2,
              float* __restrict__ accum) {
  __shared__ float sg[256], ss[256];
  int i = blockIdx.x * blockDim.x + threadIdx.x;
  float g = 0.f, sp = 0.f;
  if (i < HW) {
    int gi = top1[i];
    const float* km = keys + (size_t)gi * D;
    float s = 0.f;
    for (int d = 0; d < D; ++d) {
      float dv = qn[(size_t)d*HW + i] - km[d];
      s += dv * dv;
    }
    g = s;
    // reference's top2.T.reshape(N,2) index scramble
    const int Nh = HW / 2;
    int a, b;
    if (i < Nh) { a = top1[2*i];           b = top1[2*i + 1]; }
    else        { a = top2[2*(i - Nh)];    b = top2[2*(i - Nh) + 1]; }
    const float* ka = keys + (size_t)a * D;
    const float* kb = keys + (size_t)b * D;
    float dp = 0.f, dn = 0.f;
    for (int d = 0; d < D; ++d) {
      float q = qn[(size_t)d*HW + i];
      float e1 = q - ka[d] + 1e-6f; dp += e1*e1;
      float e2 = q - kb[d] + 1e-6f; dn += e2*e2;
    }
    sp = fmaxf(sqrtf(dp) - sqrtf(dn) + 1.0f, 0.f);
  }
  sg[threadIdx.x] = g; ss[threadIdx.x] = sp;
  __syncthreads();
  for (int s = 128; s > 0; s >>= 1) {
    if (threadIdx.x < s) {
      sg[threadIdx.x] += sg[threadIdx.x + s];
      ss[threadIdx.x] += ss[threadIdx.x + s];
    }
    __syncthreads();
  }
  if (threadIdx.x == 0) { atomicAdd(&accum[0], sg[0]); atomicAdd(&accum[1], ss[0]); }
}

__global__ void k_final(const float* accum, float* out_losses) {
  out_losses[0] = accum[0] / (float)((size_t)HW * D);   // gathering_loss
  out_losses[1] = accum[1] / (float)HW;                 // spreading_loss
}

// ---------------- weight packing: (O,I,3,3) fp32 -> (tap,O,I) bf16 ----------------
__global__ void k_pack(const float* __restrict__ w, unsigned short* __restrict__ wp,
                       int O, int I) {
  int idx = blockIdx.x * blockDim.x + threadIdx.x;
  int total = O * I * 9;
  if (idx >= total) return;
  int o = idx / (I * 9);
  int rem = idx - o * I * 9;
  int i = rem / 9;
  int t = rem - i * 9;
  wp[((size_t)t * O + o) * I + i] = f2bf(w[idx]);
}

// ---------------- conv3x3 as WMMA GEMM: y[p,co] = sum_tap,ci w[tap,co,ci]*x[p',ci]
template <int CIN, int COUT, bool RELU, bool OUTF32>
__global__ __launch_bounds__(256)
void k_conv3x3(const unsigned short* __restrict__ x,    // (HW, CIN) bf16
               const unsigned short* __restrict__ w9,   // (9, COUT, CIN) bf16
               unsigned short* __restrict__ ybf,        // (HW, COUT) bf16
               float* __restrict__ yf) {                // (HW, COUT) f32
  const int lane = threadIdx.x & 31;
  const int wave = threadIdx.x >> 5;
  const int p0 = (blockIdx.x * 8 + wave) * 32;          // 32 pixels / wave
  if (p0 >= HW) return;
  const int co0 = blockIdx.y * 64;                      // 64 out-channels / wave
  const int ln = lane & 15;
  const int kbase = (lane >> 4) * 8;

  int pp[2], py[2], px[2]; bool pv[2];
  #pragma unroll
  for (int s = 0; s < 2; ++s) {
    pp[s] = p0 + s * 16 + ln;
    pv[s] = pp[s] < HW;
    py[s] = pp[s] / W;
    px[s] = pp[s] - py[s] * W;
  }

  v8f acc[4][2];
  #pragma unroll
  for (int mt = 0; mt < 4; ++mt)
    #pragma unroll
    for (int s = 0; s < 2; ++s) acc[mt][s] = vzero();

  for (int tap = 0; tap < 9; ++tap) {
    const int dy = tap / 3 - 1, dx = tap % 3 - 1;
    const unsigned short* wt = w9 + (size_t)tap * COUT * CIN;
    int qidx[2]; bool qv[2];
    #pragma unroll
    for (int s = 0; s < 2; ++s) {
      int yy = py[s] + dy, xx = px[s] + dx;
      qv[s] = pv[s] && yy >= 0 && yy < H && xx >= 0 && xx < W;
      qidx[s] = yy * W + xx;
    }
    for (int cib = 0; cib < CIN; cib += 32) {
      Frag B[2];
      #pragma unroll
      for (int s = 0; s < 2; ++s) {
        if (qv[s]) {
          const unsigned short* r = x + (size_t)qidx[s] * CIN + cib + kbase;
          B[s].q[0] = *(const uint4*)(r);
          B[s].q[1] = *(const uint4*)(r + 16);
        } else {
          B[s].q[0] = make_uint4(0,0,0,0);
          B[s].q[1] = make_uint4(0,0,0,0);
        }
      }
      #pragma unroll
      for (int mt = 0; mt < 4; ++mt) {
        const unsigned short* r = wt + (size_t)(co0 + mt * 16 + ln) * CIN + cib + kbase;
        Frag A;
        A.q[0] = *(const uint4*)(r);
        A.q[1] = *(const uint4*)(r + 16);
        #pragma unroll
        for (int s = 0; s < 2; ++s)
          acc[mt][s] = wmma_bf16(A, B[s], acc[mt][s]);
      }
    }
  }

  const int mrow = 8 * (lane >> 4);
  #pragma unroll
  for (int s = 0; s < 2; ++s) {
    int p = p0 + s * 16 + ln;
    if (p >= HW) continue;
    #pragma unroll
    for (int mt = 0; mt < 4; ++mt) {
      int cob = co0 + mt * 16 + mrow;
      float v[8];
      #pragma unroll
      for (int r = 0; r < 8; ++r) {
        float t = acc[mt][s][r];
        v[r] = RELU ? fmaxf(t, 0.f) : t;
      }
      if (OUTF32) {
        float4* o = (float4*)&yf[(size_t)p * COUT + cob];
        o[0] = make_float4(v[0], v[1], v[2], v[3]);
        o[1] = make_float4(v[4], v[5], v[6], v[7]);
      } else {
        uint4 u;
        u.x = (unsigned)f2bf(v[0]) | ((unsigned)f2bf(v[1]) << 16);
        u.y = (unsigned)f2bf(v[2]) | ((unsigned)f2bf(v[3]) << 16);
        u.z = (unsigned)f2bf(v[4]) | ((unsigned)f2bf(v[5]) << 16);
        u.w = (unsigned)f2bf(v[6]) | ((unsigned)f2bf(v[7]) << 16);
        *(uint4*)&ybf[(size_t)p * COUT + cob] = u;
      }
    }
  }
}

// ---------------- attenmask * qn -> cfeature ----------------
__global__ void k_attn(const float* __restrict__ a, const float* __restrict__ b,
                       const float* __restrict__ qn, float* __restrict__ cf) {
  int p = blockIdx.x * blockDim.x + threadIdx.x;
  if (p >= HW) return;
  float num = 0.f, na = 0.f, nb = 0.f;
  for (int c = 0; c < 64; ++c) {
    float av = a[(size_t)p*64 + c], bv = b[(size_t)p*64 + c];
    num += av*bv; na += av*av; nb += bv*bv;
  }
  float mask = num / fmaxf(sqrtf(na) * sqrtf(nb), 1e-6f);
  for (int c = 0; c < D; ++c)
    cf[(size_t)c*HW + p] = mask * qn[(size_t)c*HW + p];
}

// ---------------- launch ----------------
extern "C" void kernel_launch(void* const* d_in, const int* in_sizes, int n_in,
                              void* d_out, int out_size, void* d_ws, size_t ws_size,
                              hipStream_t stream) {
  (void)in_sizes; (void)n_in; (void)out_size; (void)ws_size;
  const float* query = (const float*)d_in[0];
  const float* keys  = (const float*)d_in[1];
  const float* tw[4] = {(const float*)d_in[2], (const float*)d_in[3],
                        (const float*)d_in[4], (const float*)d_in[5]};
  const float* kw[4] = {(const float*)d_in[6], (const float*)d_in[7],
                        (const float*)d_in[8], (const float*)d_in[9]};

  float* out       = (float*)d_out;
  float* out_keys  = out;                              // 2048*128
  float* out_cf    = out + (size_t)M * D;              // 128*148*300
  float* out_loss  = out_cf + (size_t)D * HW;          // 2 scalars

  char* ws = (char*)d_ws;
  size_t off = 0;
  auto take = [&](size_t bytes) -> char* {
    char* p = ws + off;
    off = (off + bytes + 255) & ~(size_t)255;
    return p;
  };
  unsigned short* q2d_bf = (unsigned short*)take((size_t)HW * D * 2);  // also conv x0
  unsigned short* kn_bf  = (unsigned short*)take((size_t)M * D * 2);
  float* qn     = (float*)take((size_t)D * HW * 4);
  float* rsum   = (float*)take((size_t)M * NCHUNKS * 4);
  float* bias   = (float*)take((size_t)M * 4);
  int*   top1   = (int*)take((size_t)HW * 4);
  int*   top2   = (int*)take((size_t)HW * 4);
  float* accum  = (float*)take(64);
  unsigned short* buf512 = (unsigned short*)take((size_t)HW * 512 * 2);
  unsigned short* buf256 = (unsigned short*)take((size_t)HW * 256 * 2);
  unsigned short* buf128 = (unsigned short*)take((size_t)HW * 128 * 2);
  float* outA = (float*)take((size_t)HW * 64 * 4);
  float* outB = (float*)take((size_t)HW * 64 * 4);
  const int WO[4] = {512, 256, 128, 64};
  const int WI[4] = {128, 512, 256, 128};
  unsigned short* wp[8];
  for (int j = 0; j < 4; ++j) wp[j]     = (unsigned short*)take((size_t)9 * WO[j] * WI[j] * 2);
  for (int j = 0; j < 4; ++j) wp[4 + j] = (unsigned short*)take((size_t)9 * WO[j] * WI[j] * 2);

  // 0) zero loss accumulators
  k_init<<<1, 32, 0, stream>>>(accum);

  // 1) pack weights (t_w -> wp[0..3], k_w -> wp[4..7])
  for (int j = 0; j < 4; ++j) {
    int total = WO[j] * WI[j] * 9;
    k_pack<<<(total + 255) / 256, 256, 0, stream>>>(tw[j], wp[j],     WO[j], WI[j]);
    k_pack<<<(total + 255) / 256, 256, 0, stream>>>(kw[j], wp[4 + j], WO[j], WI[j]);
  }

  // 2) normalize
  k_norm_query<<<(HW + 255) / 256, 256, 0, stream>>>(query, qn, q2d_bf);
  k_norm_keys<<<(M + 255) / 256, 256, 0, stream>>>(keys, kn_bf, out_keys);

  // 3) fused score: row stats -> bias -> top2
  k_score_stats<<<dim3(NCHUNKS, M / 64), 128, 0, stream>>>(kn_bf, q2d_bf, rsum);
  k_bias<<<(M + 255) / 256, 256, 0, stream>>>(rsum, bias);
  k_top2<<<(HW / 16 + 3) / 4, 128, 0, stream>>>(kn_bf, q2d_bf, bias, top1, top2);

  // 4) losses
  k_losses<<<(HW + 255) / 256, 256, 0, stream>>>(qn, keys, top1, top2, accum);
  k_final<<<1, 1, 0, stream>>>(accum, out_loss);

  // 5) theta_net(qn, t_w) -> outB
  k_conv3x3<128, 512, true,  false><<<dim3(PTILES, 8), 256, 0, stream>>>(q2d_bf, wp[0], buf512, nullptr);
  k_conv3x3<512, 256, true,  false><<<dim3(PTILES, 4), 256, 0, stream>>>(buf512, wp[1], buf256, nullptr);
  k_conv3x3<256, 128, true,  false><<<dim3(PTILES, 2), 256, 0, stream>>>(buf256, wp[2], buf128, nullptr);
  k_conv3x3<128, 64,  false, true ><<<dim3(PTILES, 1), 256, 0, stream>>>(buf128, wp[3], nullptr, outB);

  // 6) theta_net(qn, k_w) -> outA
  k_conv3x3<128, 512, true,  false><<<dim3(PTILES, 8), 256, 0, stream>>>(q2d_bf, wp[4], buf512, nullptr);
  k_conv3x3<512, 256, true,  false><<<dim3(PTILES, 4), 256, 0, stream>>>(buf512, wp[5], buf256, nullptr);
  k_conv3x3<256, 128, true,  false><<<dim3(PTILES, 2), 256, 0, stream>>>(buf256, wp[6], buf128, nullptr);
  k_conv3x3<128, 64,  false, true ><<<dim3(PTILES, 1), 256, 0, stream>>>(buf128, wp[7], nullptr, outA);

  // 7) attenmask * qn -> cfeature
  k_attn<<<(HW + 255) / 256, 256, 0, stream>>>(outA, outB, qn, out_cf);
}
